// AnmlLoss_24541443129911
// MI455X (gfx1250) — matrix operand order repaired and checked by reading, not compiled, not run on hardware
//
#include <hip/hip_runtime.h>
#include <hip/hip_bf16.h>
#include <math.h>

typedef __attribute__((ext_vector_type(16))) _Float16 v16h;
typedef __attribute__((ext_vector_type(8)))  _Float16 v8h;
typedef __attribute__((ext_vector_type(8)))  float    v8f;

#define NB    4096          // batch
#define DIM   1024          // feature dim
#define MARGIN 0.09f
#define SCALE_POS 2.0f
#define SCALE_NEG 40.0f
#define EPSV 1e-5f
#define THRESH1 0.501f
#define THRESH2 0.531f

// ---------------------------------------------------------------------------
// K0: f32 -> f16 conversion of feats (values are cosine-scale, fit f16 fine)
// ---------------------------------------------------------------------------
__global__ __launch_bounds__(256) void cvt_f32_f16(const float* __restrict__ in,
                                                   _Float16* __restrict__ out) {
    int i = (blockIdx.x * 256 + threadIdx.x) * 4;
    float4 v = *(const float4*)(in + i);
    out[i + 0] = (_Float16)v.x;
    out[i + 1] = (_Float16)v.y;
    out[i + 2] = (_Float16)v.z;
    out[i + 3] = (_Float16)v.w;
}

// ---------------------------------------------------------------------------
// K1: sim = Fh * Fh^T via v_wmma_f32_16x16x32_f16.
// Block = 256 threads = 8 waves (4 row-waves x 2 col-waves).
// Wave computes a 64x64 tile = 4x4 fragments of 16x16; block tile 256x128.
// ---------------------------------------------------------------------------
__global__ __launch_bounds__(256) void sim_gemm(const _Float16* __restrict__ F,
                                                float* __restrict__ S) {
    const int lane = threadIdx.x & 31;
    const int wave = threadIdx.x >> 5;
    const int lg   = lane >> 4;    // lane group (0: lanes 0-15, 1: lanes 16-31)
    const int ln   = lane & 15;
    const int wm   = wave >> 1;    // 0..3
    const int wn   = wave & 1;     // 0..1

    const int rowBase = blockIdx.y * 256 + wm * 64;
    const int colBase = blockIdx.x * 128 + wn * 64;

    v8f acc[4][4];
#pragma unroll
    for (int m = 0; m < 4; ++m)
#pragma unroll
        for (int n = 0; n < 4; ++n)
            acc[m][n] = {};

    // Per-lane base pointers (contiguous k = lg*16 + 0..15 per lane, same for A and B)
    const _Float16* aBase[4];
    const _Float16* bBase[4];
#pragma unroll
    for (int m = 0; m < 4; ++m)
        aBase[m] = F + (size_t)(rowBase + m * 16 + ln) * DIM + lg * 16;
#pragma unroll
    for (int n = 0; n < 4; ++n)
        bBase[n] = F + (size_t)(colBase + n * 16 + ln) * DIM + lg * 16;

    for (int kb = 0; kb < DIM; kb += 32) {
        v16h afrag[4], bfrag[4];
#pragma unroll
        for (int m = 0; m < 4; ++m) {
            const _Float16* pa = aBase[m] + kb;
            v8h a0 = *(const v8h*)(pa);        // K = kb + lg*16 + 0..7
            v8h a1 = *(const v8h*)(pa + 8);    // K = kb + lg*16 + 8..15
            afrag[m] = __builtin_shufflevector(a0, a1,
                0, 1, 2, 3, 4, 5, 6, 7, 8, 9, 10, 11, 12, 13, 14, 15);
            __builtin_prefetch(pa + 64, 0, 3); // next-next K block (gfx1250 global_prefetch_b8)
        }
#pragma unroll
        for (int n = 0; n < 4; ++n) {
            const _Float16* pb = bBase[n] + kb; // K = kb + lg*16 + 0..15 contiguous
            bfrag[n] = *(const v16h*)(pb);
        }
#pragma unroll
        for (int m = 0; m < 4; ++m)
#pragma unroll
            for (int n = 0; n < 4; ++n)
                acc[m][n] = __builtin_amdgcn_wmma_f32_16x16x32_f16(
                    /*neg_a=*/false, afrag[m],
                    /*neg_b=*/false, bfrag[n],
                    /*c_mod=*/(short)0, acc[m][n],
                    /*reuse_a=*/false, /*reuse_b=*/false);
    }

    // C/D layout: VGPR v -> row (v + 8*lg), col = ln
#pragma unroll
    for (int m = 0; m < 4; ++m)
#pragma unroll
        for (int n = 0; n < 4; ++n) {
            float* p = S + (size_t)(rowBase + m * 16 + lg * 8) * NB
                         + (colBase + n * 16 + ln);
#pragma unroll
            for (int v = 0; v < 8; ++v)
                p[(size_t)v * NB] = acc[m][n][v];
        }
}

// ---------------------------------------------------------------------------
// K2: per-row max over negatives (labels differ). One block per row.
// ---------------------------------------------------------------------------
__global__ __launch_bounds__(256) void row_maxneg(const float* __restrict__ S,
                                                  const long long* __restrict__ labels,
                                                  float* __restrict__ maxneg) {
    const int r = blockIdx.x;
    const long long lr = labels[r];
    const float* row = S + (size_t)r * NB;
    float m = -INFINITY;
    for (int j = threadIdx.x; j < NB; j += 256)
        if (labels[j] != lr) m = fmaxf(m, row[j]);
    __shared__ float red[256];
    red[threadIdx.x] = m;
    __syncthreads();
    for (int s = 128; s > 0; s >>= 1) {
        if (threadIdx.x < s)
            red[threadIdx.x] = fmaxf(red[threadIdx.x], red[threadIdx.x + s]);
        __syncthreads();
    }
    if (threadIdx.x == 0) maxneg[r] = red[0];
}

// ---------------------------------------------------------------------------
// K3: per-row masked exp-sums + per-row loss. One block per row.
// ---------------------------------------------------------------------------
__global__ __launch_bounds__(256) void row_loss(const float* __restrict__ S,
                                                const long long* __restrict__ labels,
                                                const float* __restrict__ maxneg,
                                                float* __restrict__ rowloss) {
    const int r = blockIdx.x;
    const long long lr = labels[r];
    const float mn = maxneg[r];
    const float* row = S + (size_t)r * NB;

    float n_pos = 0.f, n_neg = 0.f, pos_sum = 0.f, neg_sum = 0.f;
    for (int j = threadIdx.x; j < NB; j += 256) {
        float s = row[j];
        if (labels[j] != lr) {
            n_neg += 1.f;
            neg_sum += __expf(SCALE_NEG * s);
        } else if (j != r && s < (1.f - EPSV) && (s - MARGIN) < mn) {
            n_pos += 1.f;
            pos_sum += __expf(-SCALE_POS * s);
        }
    }

    __shared__ float rnp[256], rnn[256], rps[256], rns[256];
    rnp[threadIdx.x] = n_pos; rnn[threadIdx.x] = n_neg;
    rps[threadIdx.x] = pos_sum; rns[threadIdx.x] = neg_sum;
    __syncthreads();
    for (int s = 128; s > 0; s >>= 1) {
        if (threadIdx.x < s) {
            rnp[threadIdx.x] += rnp[threadIdx.x + s];
            rnn[threadIdx.x] += rnn[threadIdx.x + s];
            rps[threadIdx.x] += rps[threadIdx.x + s];
            rns[threadIdx.x] += rns[threadIdx.x + s];
        }
        __syncthreads();
    }
    if (threadIdx.x == 0) {
        float np = rnp[0], nn = rnn[0], ps = rps[0], ns = rns[0];
        float pos_loss = (1.f / SCALE_POS) *
            logf((ps + __expf(-SCALE_POS * THRESH1)) / (np + 1.f));
        float neg_loss = (1.f / SCALE_NEG) *
            logf((ns + __expf(SCALE_NEG * THRESH2)) / (nn + 1.f));
        float pr = logf(5.33f + __expf(pos_loss + neg_loss));
        rowloss[r] = (nn >= 1.f && np >= 1.f) ? pr : 0.f;
    }
}

// ---------------------------------------------------------------------------
// K4: deterministic final mean over 4096 row losses.
// ---------------------------------------------------------------------------
__global__ __launch_bounds__(256) void final_mean(const float* __restrict__ rowloss,
                                                  float* __restrict__ out) {
    float s = 0.f;
    for (int i = threadIdx.x; i < NB; i += 256) s += rowloss[i];
    __shared__ float red[256];
    red[threadIdx.x] = s;
    __syncthreads();
    for (int t = 128; t > 0; t >>= 1) {
        if (threadIdx.x < t) red[threadIdx.x] += red[threadIdx.x + t];
        __syncthreads();
    }
    if (threadIdx.x == 0) out[0] = red[0] / (float)NB;
}

// ---------------------------------------------------------------------------
extern "C" void kernel_launch(void* const* d_in, const int* in_sizes, int n_in,
                              void* d_out, int out_size, void* d_ws, size_t ws_size,
                              hipStream_t stream) {
    const float*     feats  = (const float*)d_in[0];
    const long long* labels = (const long long*)d_in[1];   // int64 labels
    float*           out    = (float*)d_out;

    char* ws = (char*)d_ws;
    _Float16* Fh      = (_Float16*)ws;                                   //  8 MB
    float*    S       = (float*)(ws + (size_t)8  * 1024 * 1024);         // 64 MB
    float*    maxneg  = (float*)(ws + (size_t)72 * 1024 * 1024);         // 16 KB
    float*    rowloss = maxneg + NB;                                     // 16 KB

    // K0: convert feats to f16 (4M elements, 4 per thread)
    cvt_f32_f16<<<(NB * DIM) / (256 * 4), 256, 0, stream>>>(feats, Fh);

    // K1: WMMA GEMM sim = Fh * Fh^T (grid: 32 col-blocks x 16 row-blocks)
    dim3 ggrid(NB / 128, NB / 256);
    sim_gemm<<<ggrid, 256, 0, stream>>>(Fh, S);

    // K2: per-row max over negatives
    row_maxneg<<<NB, 256, 0, stream>>>(S, labels, maxneg);

    // K3: per-row masked sums + loss
    row_loss<<<NB, 256, 0, stream>>>(S, labels, maxneg, rowloss);

    // K4: mean
    final_mean<<<1, 256, 0, stream>>>(rowloss, out);
}
